// DistAttention_23742579212704
// MI455X (gfx1250) — matrix operand order repaired and checked by reading, not compiled, not run on hardware
//
#include <hip/hip_runtime.h>

// ---------- types ----------
typedef __bf16 bf16_t;
typedef bf16_t v16bf __attribute__((ext_vector_type(16)));
typedef bf16_t v8bf  __attribute__((ext_vector_type(8)));
typedef float  v8f   __attribute__((ext_vector_type(8)));

union BF16x16 { v16bf v; v8bf h[2]; };

#define NUM_HEAD 16
#define HEAD_DIM 64
#define HIDDEN   1024
#define N3       3072
#define BATCH    2
#define SEQ      2048
#define MROWS    (BATCH * SEQ)   // 4096

// ---------- kernel 1: x f32 -> bf16 (same layout) ----------
__global__ void cvt_x_kernel(const float* __restrict__ x, bf16_t* __restrict__ xb, int n) {
    int i = blockIdx.x * blockDim.x + threadIdx.x;
    if (i < n) xb[i] = (bf16_t)x[i];
}

// ---------- kernel 2: W f32 (K x N) -> bf16 transposed Wt (N x K) ----------
__global__ void cvt_w_kernel(const float* __restrict__ W, bf16_t* __restrict__ Wt, int n) {
    int i = blockIdx.x * blockDim.x + threadIdx.x;
    if (i < n) {
        int k = i / N3;
        int c = i - k * N3;
        Wt[(size_t)c * HIDDEN + k] = (bf16_t)W[i];
    }
}

// ---------- kernel 3: QKV GEMM (bf16 WMMA, f32 accum) ----------
// C[4096 x 3072] = xb @ Wt^T + bias ; scatter to Q [bh][s][d], K [bh][s][d], Vt [bh][d][s]
__global__ __launch_bounds__(128) void qkv_gemm_kernel(
    const bf16_t* __restrict__ xb, const bf16_t* __restrict__ Wt,
    const float* __restrict__ bias,
    bf16_t* __restrict__ qb, bf16_t* __restrict__ kb, bf16_t* __restrict__ vt)
{
    const int lane = threadIdx.x & 31;
    const int wave = threadIdx.x >> 5;
    const int l15  = lane & 15;
    const int g    = lane >> 4;
    const int koffA = g ? 16 : 0;   // A frag: group1 holds K 16..31 contiguous
    const int koffB = g ? 16 : 0;   // B frag: group1 holds K 16..31 contiguous

    const int n0 = blockIdx.x * 64;
    const int m0 = blockIdx.y * 128 + wave * 32;

    v8f vz = {0.f,0.f,0.f,0.f,0.f,0.f,0.f,0.f};
    v8f acc[2][4];
    #pragma unroll
    for (int mb = 0; mb < 2; mb++)
        #pragma unroll
        for (int nb = 0; nb < 4; nb++) acc[mb][nb] = vz;

    for (int k = 0; k < HIDDEN; k += 32) {
        BF16x16 a[2];
        #pragma unroll
        for (int mb = 0; mb < 2; mb++) {
            const bf16_t* p = xb + (size_t)(m0 + mb*16 + l15) * HIDDEN + k + koffA;
            a[mb].h[0] = *(const v8bf*)p;
            a[mb].h[1] = *(const v8bf*)(p + 8);
        }
        #pragma unroll
        for (int nb = 0; nb < 4; nb++) {
            BF16x16 bm;
            const bf16_t* p = Wt + (size_t)(n0 + nb*16 + l15) * HIDDEN + k + koffB;
            bm.h[0] = *(const v8bf*)p;
            bm.h[1] = *(const v8bf*)(p + 8);
            #pragma unroll
            for (int mb = 0; mb < 2; mb++) {
                acc[mb][nb] = __builtin_amdgcn_wmma_f32_16x16x32_bf16(
                    false, a[mb].v, false, bm.v, (short)0, acc[mb][nb], false, false);
            }
        }
    }

    // epilogue: + bias, convert to bf16, scatter to Q/K/Vt
    #pragma unroll
    for (int nb = 0; nb < 4; nb++) {
        int c = n0 + nb*16 + l15;
        float bv = bias[c];
        int which = c >> 10;            // 0=q 1=k 2=v
        int hh = (c & 1023) >> 6;
        int dd = c & 63;
        #pragma unroll
        for (int mb = 0; mb < 2; mb++) {
            #pragma unroll
            for (int i = 0; i < 8; i++) {
                int r = m0 + mb*16 + i + 8*g;
                int batch = r >> 11;
                int s = r & 2047;
                float v = acc[mb][nb][i] + bv;
                bf16_t bvv = (bf16_t)v;
                size_t bh = (size_t)(batch * NUM_HEAD + hh);
                if (which == 0)      qb[(bh * SEQ + s) * HEAD_DIM + dd] = bvv;
                else if (which == 1) kb[(bh * SEQ + s) * HEAD_DIM + dd] = bvv;
                else                 vt[(bh * HEAD_DIM + dd) * SEQ + s] = bvv;
            }
        }
    }
}

// ---------- kernel 4: flash attention, 1 wave per 16-row Q block ----------
__global__ __launch_bounds__(128) void attn_kernel(
    const bf16_t* __restrict__ qg, const bf16_t* __restrict__ kg,
    const bf16_t* __restrict__ vtg, float* __restrict__ out)
{
    __shared__ bf16_t plds[4][16][32];   // per-wave P transpose bounce

    const int lane = threadIdx.x & 31;
    const int wave = threadIdx.x >> 5;
    const int l15  = lane & 15;
    const int g    = lane >> 4;
    const int koffA = g ? 16 : 0;
    const int koffB = g ? 16 : 0;

    int rb    = blockIdx.x * 4 + wave;   // 0..4095 row blocks
    int sblk  = rb & (SEQ/16 - 1);       // 0..127
    int bh    = rb >> 7;                 // 0..31
    int sbase = sblk * 16;
    int batch = bh >> 4;
    int head  = bh & 15;

    const bf16_t* Q = qg  + (size_t)bh * SEQ * HEAD_DIM;
    const bf16_t* K = kg  + (size_t)bh * SEQ * HEAD_DIM;
    const bf16_t* V = vtg + (size_t)bh * HEAD_DIM * SEQ;

    // Q A-fragments (rows sbase..sbase+15, head dim split into two 32-chunks)
    BF16x16 aq[2];
    #pragma unroll
    for (int ks = 0; ks < 2; ks++) {
        const bf16_t* p = Q + (size_t)(sbase + l15) * HEAD_DIM + ks*32 + koffA;
        aq[ks].h[0] = *(const v8bf*)p;
        aq[ks].h[1] = *(const v8bf*)(p + 8);
    }

    v8f vz = {0.f,0.f,0.f,0.f,0.f,0.f,0.f,0.f};
    float mrun[8], lrun[8];
    v8f o[4];
    #pragma unroll
    for (int i = 0; i < 8; i++) { mrun[i] = -3.0e38f; lrun[i] = 0.f; }
    #pragma unroll
    for (int nb = 0; nb < 4; nb++) o[nb] = vz;

    const float scale = 0.125f;  // 1/sqrt(64)

    for (int tb = 0; tb <= sbase + 15; tb += 32) {
        // ---- scores = Q @ K^T for this 16x32 block ----
        v8f sc[2]; sc[0] = vz; sc[1] = vz;
        #pragma unroll
        for (int nb = 0; nb < 2; nb++) {
            #pragma unroll
            for (int ks = 0; ks < 2; ks++) {
                BF16x16 bk;
                const bf16_t* p = K + (size_t)(tb + nb*16 + l15) * HEAD_DIM + ks*32 + koffB;
                bk.h[0] = *(const v8bf*)p;
                bk.h[1] = *(const v8bf*)(p + 8);
                sc[nb] = __builtin_amdgcn_wmma_f32_16x16x32_bf16(
                    false, aq[ks].v, false, bk.v, (short)0, sc[nb], false, false);
            }
        }
        // ---- scale + causal mask ----
        #pragma unroll
        for (int nb = 0; nb < 2; nb++) {
            int t = tb + nb*16 + l15;
            #pragma unroll
            for (int i = 0; i < 8; i++) {
                int r = sbase + i + 8*g;
                float s = sc[nb][i] * scale;
                sc[nb][i] = (t > r) ? -1.0e30f : s;
            }
        }
        // ---- online softmax: row max / exp / row sum (reduce across 16-lane half) ----
        float mnew[8], alpha[8];
        #pragma unroll
        for (int i = 0; i < 8; i++) {
            float v = fmaxf(sc[0][i], sc[1][i]);
            v = fmaxf(v, __shfl_xor(v, 1, 32));
            v = fmaxf(v, __shfl_xor(v, 2, 32));
            v = fmaxf(v, __shfl_xor(v, 4, 32));
            v = fmaxf(v, __shfl_xor(v, 8, 32));
            mnew[i]  = fmaxf(mrun[i], v);
            alpha[i] = __expf(mrun[i] - mnew[i]);
        }
        #pragma unroll
        for (int nb = 0; nb < 2; nb++)
            #pragma unroll
            for (int i = 0; i < 8; i++)
                sc[nb][i] = __expf(sc[nb][i] - mnew[i]);
        #pragma unroll
        for (int i = 0; i < 8; i++) {
            float v = sc[0][i] + sc[1][i];
            v += __shfl_xor(v, 1, 32);
            v += __shfl_xor(v, 2, 32);
            v += __shfl_xor(v, 4, 32);
            v += __shfl_xor(v, 8, 32);
            lrun[i] = lrun[i] * alpha[i] + v;
            mrun[i] = mnew[i];
        }
        // ---- rescale running output ----
        #pragma unroll
        for (int nb = 0; nb < 4; nb++)
            #pragma unroll
            for (int i = 0; i < 8; i++)
                o[nb][i] *= alpha[i];
        // ---- P (C-layout) -> LDS -> A-fragment ----
        #pragma unroll
        for (int nb = 0; nb < 2; nb++)
            #pragma unroll
            for (int i = 0; i < 8; i++)
                plds[wave][i + 8*g][nb*16 + l15] = (bf16_t)sc[nb][i];
        BF16x16 ap;
        {
            const bf16_t* p = &plds[wave][l15][koffA];
            ap.h[0] = *(const v8bf*)p;
            ap.h[1] = *(const v8bf*)(p + 8);
        }
        // ---- o += P @ V (V stored transposed: [d][s]) ----
        #pragma unroll
        for (int nb = 0; nb < 4; nb++) {
            BF16x16 bv;
            const bf16_t* p = V + (size_t)(nb*16 + l15) * SEQ + tb + koffB;
            bv.h[0] = *(const v8bf*)p;
            bv.h[1] = *(const v8bf*)(p + 8);
            o[nb] = __builtin_amdgcn_wmma_f32_16x16x32_bf16(
                false, ap.v, false, bv.v, (short)0, o[nb], false, false);
        }
    }

    // ---- write out (b, s, h, d) f32 ----
    #pragma unroll
    for (int nb = 0; nb < 4; nb++) {
        int d = nb*16 + l15;
        #pragma unroll
        for (int i = 0; i < 8; i++) {
            int s = sbase + i + 8*g;
            float v = o[nb][i] / lrun[i];
            out[(((size_t)batch * SEQ + s) * NUM_HEAD + head) * HEAD_DIM + d] = v;
        }
    }
}

// ---------- launch ----------
extern "C" void kernel_launch(void* const* d_in, const int* in_sizes, int n_in,
                              void* d_out, int out_size, void* d_ws, size_t ws_size,
                              hipStream_t stream) {
    const float* x    = (const float*)d_in[0];
    const float* W    = (const float*)d_in[1];
    const float* bias = (const float*)d_in[2];
    float* out = (float*)d_out;

    char* ws = (char*)d_ws;
    const size_t MB = 1024u * 1024u;
    bf16_t* xb = (bf16_t*)(ws);             //  8 MB : 4096x1024 bf16
    bf16_t* Wt = (bf16_t*)(ws +  8 * MB);   //  6 MB : 3072x1024 bf16 (transposed)
    bf16_t* qb = (bf16_t*)(ws + 14 * MB);   //  8 MB : [bh][s][d]
    bf16_t* kb = (bf16_t*)(ws + 22 * MB);   //  8 MB : [bh][s][d]
    bf16_t* vt = (bf16_t*)(ws + 30 * MB);   //  8 MB : [bh][d][s]

    const int nx = MROWS * HIDDEN;   // 4,194,304
    const int nw = HIDDEN * N3;      // 3,145,728

    cvt_x_kernel<<<(nx + 255) / 256, 256, 0, stream>>>(x, xb, nx);
    cvt_w_kernel<<<(nw + 255) / 256, 256, 0, stream>>>(W, Wt, nw);

    dim3 gg(N3 / 64, MROWS / 128);   // (48, 32)
    qkv_gemm_kernel<<<gg, 128, 0, stream>>>(xb, Wt, bias, qb, kb, vt);

    int nblocks = (BATCH * NUM_HEAD * (SEQ / 16)) / 4;  // 1024
    attn_kernel<<<nblocks, 128, 0, stream>>>(qb, kb, vt, out);
}
